// SynthesisLayer_73830487818668
// MI455X (gfx1250) — matrix operand order repaired and checked by reading, not compiled, not run on hardware
//
#include <hip/hip_runtime.h>

// ---------------------------------------------------------------------------
// SynthesisLayer for MI455X (gfx1250, wave32, WMMA).
//
// Problem:  B=4, NUM_FACE=49152 (BN=196608), CIN=COUT=64, K2=9, WDIM=512,
//           PAD=4096.  Core op is a gathered GEMM: M=faces, N=64, K=9*64=576.
// Roofline: 14.5 GFLOP vs ~100 MB HBM -> f16 WMMA w/ f32 accumulate, gather
//           served out of L2 (x fits: 25 MB f16 table << 192 MB L2).
// This rev: preload all 4 B operands per K-tile into distinct registers so
// the 8 ds_load_b128 clause together and overlap the 8 WMMAs (1 DS wait per
// K-tile instead of 4 full drains).
// ---------------------------------------------------------------------------

#define NBATCH   4
#define NFACE    49152
#define BNROWS   196608        // NBATCH * NFACE
#define NPAD     4096
#define CIN      64
#define COUT     64
#define K2       9
#define WDIM     512
#define KTOT     576           // K2 * CIN
#define KTILES   18            // KTOT / 32

typedef __attribute__((ext_vector_type(4)))  _Float16 v4h;
typedef __attribute__((ext_vector_type(8)))  _Float16 v8h;   // 16 bytes
typedef __attribute__((ext_vector_type(16))) _Float16 v16h;  // WMMA A/B operand
typedef __attribute__((ext_vector_type(8)))  float    v8f;   // WMMA C/D operand

union V16 { v16h v; v8h h[2]; };

// ---------------------------------------------------------------------------
// styles[b][i] = (w[b,:] . affine_w[i,:]) / sqrt(512) + affine_b[i]
// ---------------------------------------------------------------------------
__global__ void styles_kernel(const float* __restrict__ w,
                              const float* __restrict__ affine_w,
                              const float* __restrict__ affine_b,
                              float* __restrict__ styles) {
  const int tid = threadIdx.x;            // 256 threads, 1 block
  const int b = tid >> 6, i = tid & 63;
  const float gain = 0.04419417382415922f;  // 1/sqrt(512)
  const float* wr = w + b * WDIM;
  const float* ar = affine_w + i * WDIM;
  float acc = 0.f;
  for (int d = 0; d < WDIM; ++d) acc += wr[d] * ar[d];
  styles[b * CIN + i] = acc * gain + affine_b[i];
}

// ---------------------------------------------------------------------------
// Build demodulated wmod in f16, pre-packed for the WMMA B operand.
// Reduction index K = k*64 + i  (neighbor-major, cin-minor) to match A loads.
// Bpack layout: [b][kt(18)][n=cout(64)][kk(32)]  (column-major per 32-K tile),
// so B-operand lane L reads two 8-half runs at (col=L%16, kk=(L/16)*8 and (L/16)*8+16).
// One wave per (b, o); lane handles K = j*32 + lane for j=0..17.
// ---------------------------------------------------------------------------
__global__ void pack_kernel(const float* __restrict__ weight,   // [O,I,1,K2]
                            const float* __restrict__ styles,   // [B,I]
                            _Float16* __restrict__ Bpack) {
  const int lane = threadIdx.x & 31;
  const int gw   = blockIdx.x * (blockDim.x >> 5) + (threadIdx.x >> 5); // 0..255
  const int b = gw >> 6, o = gw & 63;
  const float* st = styles + b * CIN;

  float wm[KTILES];
  float ss = 0.f;
#pragma unroll
  for (int j = 0; j < KTILES; ++j) {
    const int K = j * 32 + lane;
    const int i = K & 63, k = K >> 6;               // K = k*64 + i
    const float v = weight[(o * CIN + i) * K2 + k] * st[i];
    wm[j] = v;
    ss += v * v;
  }
  // wave32 butterfly reduction for sum of squares
#pragma unroll
  for (int m = 16; m >= 1; m >>= 1) ss += __shfl_xor(ss, m, 32);
  const float d = rsqrtf(ss + 1e-8f);
#pragma unroll
  for (int j = 0; j < KTILES; ++j)
    Bpack[(((size_t)b * KTILES + j) * COUT + o) * 32 + lane] =
        (_Float16)(wm[j] * d);
}

// ---------------------------------------------------------------------------
// x (f32) -> f16 table with (BN + PAD) rows; pad rows are zero so the
// neighbor gather needs no masking (face_is_pad is structurally first-BN-real).
// ---------------------------------------------------------------------------
__global__ void convert_kernel(const float* __restrict__ x,
                               _Float16* __restrict__ xh) {
  const int idx = (blockIdx.x * blockDim.x + threadIdx.x) * 4;  // < 12,845,056
  float4 v = {0.f, 0.f, 0.f, 0.f};
  if (idx < BNROWS * CIN) v = *reinterpret_cast<const float4*>(x + idx);
  v4h h = {(_Float16)v.x, (_Float16)v.y, (_Float16)v.z, (_Float16)v.w};
  *reinterpret_cast<v4h*>(xh + idx) = h;
}

// ---------------------------------------------------------------------------
// Main gathered GEMM.  256 threads = 8 waves; each wave owns a 32-face x
// 64-cout tile: 8 f32 accumulators (v8f), 18 K-tiles -> 144 WMMAs per wave.
// A operand: per-lane b128 gathers from the f16 x-table (row-major 16-bit
// tile layout: lane L -> row L%16, 8-half chunk (L/16); chunks at +0/+32 B),
// addressed as uniform-base + 32-bit byte offset.
// B operand: staged once per block into LDS (72 KB/batch, rows padded to
// 80 B to spread banks); all 4 cout-tile operands preloaded per K-tile so the
// 8 ds_load_b128 clause together and overlap the 8 WMMAs.
// ---------------------------------------------------------------------------
#define LROW 40   // padded LDS row stride in halfs (32 data + 8 pad = 80 B)

__global__ __launch_bounds__(256) void faceconv_kernel(
    const _Float16* __restrict__ xh,     // [(BN+PAD) x 64] f16
    const int*      __restrict__ nbrs,   // [BN x 9]
    const _Float16* __restrict__ Bpack,  // [B x 18 x 64 x 32] f16
    const float*    __restrict__ noise_const,     // [NFACE]
    const float*    __restrict__ noise_strength,  // [1]
    const float*    __restrict__ bias,            // [COUT]
    float*          __restrict__ out)             // [BN x 64] f32
{
  __shared__ __align__(16) _Float16 Bsh[KTILES * COUT * LROW];  // 92160 B

  const int tid  = threadIdx.x;
  const int wave = tid >> 5;
  const int lane = tid & 31;
  const int lo   = lane & 15;
  const int hi   = lane >> 4;

  const int tile      = blockIdx.x * 8 + wave;   // 6144 tiles of 32 faces
  const int face_base = tile * 32;
  const int b         = face_base / NFACE;       // uniform per block (192 blk/batch)
  const int n_base    = face_base % NFACE;

  // --- stage this batch's packed B into LDS (64 B data per row -> 80 B row) --
  {
    const v8h* __restrict__ src =
        reinterpret_cast<const v8h*>(Bpack + (size_t)b * KTILES * COUT * 32);
    v8h* __restrict__ dst = reinterpret_cast<v8h*>(Bsh);
#pragma unroll
    for (int j = 0; j < 18; ++j) {                 // 4608 16-B chunks / 256 thr
      const int c   = j * 256 + tid;
      const int row = c >> 2, sub = c & 3;         // row < 1152
      dst[row * 5 + sub] = src[c];                 // 5*16B = 80B padded row
    }
  }

  // --- neighbor byte-offsets for this lane's two faces (rows lo, lo+16) -----
  // off = nbr*128 bytes (row in f16 table) + hi*16 (lane's A chunk).
  const int f0 = face_base + lo;
  unsigned off0[K2], off1[K2];
#pragma unroll
  for (int k = 0; k < K2; ++k) {
    off0[k] = ((unsigned)nbrs[f0 * K2 + k]        << 7) + (hi << 4);
    off1[k] = ((unsigned)nbrs[(f0 + 16) * K2 + k] << 7) + (hi << 4);
  }

  __syncthreads();

  const char* __restrict__ xhb = reinterpret_cast<const char*>(xh);
  v8f acc0[4] = {};   // faces face_base..+15
  v8f acc1[4] = {};   // faces face_base+16..+31

#pragma unroll
  for (int kt = 0; kt < KTILES; ++kt) {
    const int k  = kt >> 1;        // neighbor
    const int hf = kt & 1;         // which 32-cin half of that neighbor's row
    const v8h* __restrict__ a0 =
        reinterpret_cast<const v8h*>(xhb + off0[k] + (hf << 6));
    const v8h* __restrict__ a1 =
        reinterpret_cast<const v8h*>(xhb + off1[k] + (hf << 6));
    V16 A0, A1;
    A0.h[0] = a0[0];  A0.h[1] = a0[2];   // K 0..7 / 16..23 (hi=0); 8..15 / 24..31 (hi=1)
    A1.h[0] = a1[0];  A1.h[1] = a1[2];

    // Preload all 4 cout-tile B operands -> one clause of 8 ds_load_b128.
    V16 Bv[4];
#pragma unroll
    for (int nt = 0; nt < 4; ++nt) {
      const v8h* __restrict__ brow = reinterpret_cast<const v8h*>(
          Bsh + (size_t)(kt * COUT + nt * 16 + lo) * LROW + hi * 8);
      Bv[nt].h[0] = brow[0];
      Bv[nt].h[1] = brow[2];
    }

#pragma unroll
    for (int nt = 0; nt < 4; ++nt) {
      acc0[nt] = __builtin_amdgcn_wmma_f32_16x16x32_f16(
          false, A0.v, false, Bv[nt].v, (short)0, acc0[nt], false, false);
      acc1[nt] = __builtin_amdgcn_wmma_f32_16x16x32_f16(
          false, A1.v, false, Bv[nt].v, (short)0, acc1[nt], false, false);
    }
  }

  // --- epilogue: noise + bias, lrelu(0.2)*sqrt(2), clamp +/-256 -------------
  const float ns = noise_strength[0];
  float nz0[8], nz1[8];
#pragma unroll
  for (int v = 0; v < 8; ++v) {
    nz0[v] = noise_const[n_base + hi * 8 + v] * ns;        // rows +0..15
    nz1[v] = noise_const[n_base + 16 + hi * 8 + v] * ns;   // rows +16..31
  }

#pragma unroll
  for (int nt = 0; nt < 4; ++nt) {
    const int col  = nt * 16 + lo;
    const float bc = bias[col];
#pragma unroll
    for (int v = 0; v < 8; ++v) {
      float y0 = acc0[nt][v] + nz0[v] + bc;
      y0 = (y0 > 0.f) ? y0 : 0.2f * y0;
      y0 *= 1.4142135623730951f;
      y0 = fminf(fmaxf(y0, -256.f), 256.f);
      out[(size_t)(face_base + hi * 8 + v) * COUT + col] = y0;

      float y1 = acc1[nt][v] + nz1[v] + bc;
      y1 = (y1 > 0.f) ? y1 : 0.2f * y1;
      y1 *= 1.4142135623730951f;
      y1 = fminf(fmaxf(y1, -256.f), 256.f);
      out[(size_t)(face_base + 16 + hi * 8 + v) * COUT + col] = y1;
    }
  }
}

// ---------------------------------------------------------------------------
extern "C" void kernel_launch(void* const* d_in, const int* in_sizes, int n_in,
                              void* d_out, int out_size, void* d_ws, size_t ws_size,
                              hipStream_t stream) {
  const float* x            = (const float*)d_in[0];
  const int*   nbrs         = (const int*)  d_in[1];
  // d_in[2] = face_is_pad: structurally first-BN-real/last-PAD-pad -> unused
  const float* w            = (const float*)d_in[3];
  const float* affine_w     = (const float*)d_in[4];
  const float* affine_b     = (const float*)d_in[5];
  const float* weight       = (const float*)d_in[6];
  const float* noise_const  = (const float*)d_in[7];
  const float* noise_str    = (const float*)d_in[8];
  const float* bias         = (const float*)d_in[9];
  float*       out          = (float*)d_out;

  char* ws = (char*)d_ws;
  float*    styles = (float*)ws;                      // 1 KB
  _Float16* Bpack  = (_Float16*)(ws + 4096);          // 294,912 B
  _Float16* xh     = (_Float16*)(ws + 303104);        // 25,690,112 B (~26 MB total)

  styles_kernel <<<1,     256, 0, stream>>>(w, affine_w, affine_b, styles);
  pack_kernel   <<<32,    256, 0, stream>>>(weight, styles, Bpack);
  convert_kernel<<<12544, 256, 0, stream>>>(x, xh);   // (BN+PAD)*64/4 threads
  faceconv_kernel<<<768,  256, 0, stream>>>(xh, nbrs, Bpack, noise_const,
                                            noise_str, bias, out);
}
